// EntityRoleClassifier_79671643341582
// MI455X (gfx1250) — matrix operand order repaired and checked by reading, not compiled, not run on hardware
//
#include <hip/hip_runtime.h>
#include <hip/hip_bf16.h>
#include <math.h>

typedef __attribute__((ext_vector_type(16))) _Float16 v16h;
typedef __attribute__((ext_vector_type(8)))  _Float16 v8h;
typedef __attribute__((ext_vector_type(8)))  float    v8f;

#define B_    64
#define S_    512
#define H_    768
#define E_    8
#define HID_  256
#define K3H   2304        /* 3*H */
#define ROWS  512         /* B*E */
#define NCH   32          /* chunks per sequence */
#define CHL   16          /* rows per chunk */

/* workspace layout (bytes), all 256B aligned.
   CS (phase A->B temp) and HID (GEMM1 output) alias: CS is dead before GEMM1. */
#define OFF_EMB   0u
#define OFF_W1T   (ROWS * K3H * 2u)                    /* 2,359,296            */
#define OFF_CS    (OFF_W1T + 4u * K3H * HID_ * 2u)     /* 7,077,888            */
#define OFF_HID   OFF_CS                               /* alias, 2MB < 6.3MB   */
/* total ws use: OFF_CS + 64*32*768*4 = 13,369,344 bytes (~12.75 MB) */

/* ---------------- kernel 0: W1 -> f16 transposed (per head) -------------- */
__global__ __launch_bounds__(256) void prep_w1_all(
    const float* __restrict__ w1m, const float* __restrict__ w1p,
    const float* __restrict__ w1a, const float* __restrict__ w1i,
    _Float16* __restrict__ w1t) {
  const int head = blockIdx.y;
  const float* w1 = head == 0 ? w1m : head == 1 ? w1p : head == 2 ? w1a : w1i;
  const int idx = blockIdx.x * 256 + threadIdx.x;   /* 0 .. 2304*256-1 */
  const int k = idx / HID_;
  const int n = idx % HID_;
  w1t[(size_t)head * K3H * HID_ + (size_t)n * K3H + k] = (_Float16)w1[idx];
}

/* ------ kernel 1a: bandwidth phase — 16-row chunk sums, 1 add/element ----- */
__global__ __launch_bounds__(256) void chunk_sums(
    const float* __restrict__ seq, float* __restrict__ cs) {
  const int idx = blockIdx.x * 256 + threadIdx.x;  /* over 64*32*768 */
  const int h  = idx % H_;
  const int bc = idx / H_;                         /* b*32 + c */
  const int b  = bc >> 5;
  const int c  = bc & (NCH - 1);
  const float* p = seq + ((size_t)(b * S_ + c * CHL)) * H_ + h;
  float acc = 0.f;
#pragma unroll
  for (int i = 0; i < CHL; ++i) acc += p[(size_t)i * H_];
  cs[idx] = acc;
}

/* ------ kernel 1b: cheap phase — prefixes from chunk sums + L2 edges ------ */
__global__ __launch_bounds__(256) void span_embed2(
    const float* __restrict__ seq, const float* __restrict__ cs,
    const int* __restrict__ starts, const int* __restrict__ ends,
    _Float16* __restrict__ emb) {
  const int b = blockIdx.y;
  const int h = blockIdx.x * 256 + threadIdx.x;    /* 0..767 */

  float csr[NCH];
  const float* cp = cs + (size_t)b * NCH * H_ + h;
#pragma unroll
  for (int c = 0; c < NCH; ++c) csr[c] = cp[(size_t)c * H_];
  float ptot = 0.f;
#pragma unroll
  for (int c = 0; c < NCH; ++c) ptot += csr[c];

  const float* sp = seq + (size_t)b * S_ * H_ + h;

#pragma unroll
  for (int j = 0; j < E_; ++j) {
    const int s  = starts[b * E_ + j];            /* block-uniform */
    const int e  = ends[b * E_ + j];
    const int e1 = e + 1;
    const int cb = s >> 4;                        /* full chunks below s */

    float psum = 0.f;                              /* P[s] */
#pragma unroll
    for (int c = 0; c < NCH; ++c) psum += (c < cb) ? csr[c] : 0.f;
    for (int t = cb << 4; t < s; ++t)              /* <=15 L2-hit edge rows */
      psum += sp[(size_t)t * H_];

    float ssum = 0.f;                              /* rows s..e, <=12 */
    for (int t = s; t <= e; ++t)
      ssum += sp[(size_t)t * H_];

    const float pe1   = psum + ssum;               /* P[e+1] */
    const float left  = (s > 0) ? psum / (float)s : 0.f;
    const float span  = ssum / (float)(e1 - s);
    const int   rc    = S_ - e1;
    const float right = (rc > 0) ? (ptot - pe1) / (float)rc : 0.f;

    const size_t row = (size_t)(b * E_ + j) * K3H;
    emb[row + h]          = (_Float16)left;
    emb[row + H_ + h]     = (_Float16)span;
    emb[row + 2 * H_ + h] = (_Float16)right;
  }
}

/* ------------- kernel 2: GEMM1 via v_wmma_f32_16x16x32_f16 ---------------- */
/* one wave computes one 16x16 tile of hidden = relu(emb @ W1 + b1)           */
__global__ __launch_bounds__(256) void mlp1_wmma(
    const _Float16* __restrict__ emb, const _Float16* __restrict__ w1t,
    const float* __restrict__ b1m, const float* __restrict__ b1p,
    const float* __restrict__ b1a, const float* __restrict__ b1i,
    float* __restrict__ hidden) {
  const int wave = (blockIdx.x * 256 + threadIdx.x) >> 5;  /* 0..2047 */
  const int lane = threadIdx.x & 31;
  const int head = wave >> 9;         /* / (32*16) */
  const int rem  = wave & 511;
  const int mt   = rem >> 4;          /* 0..31 row tiles   */
  const int nt   = rem & 15;          /* 0..15 col tiles   */
  const float* b1 = head == 0 ? b1m : head == 1 ? b1p : head == 2 ? b1a : b1i;

  const int l15 = lane & 15;
  const int hi  = lane >> 4;
  /* A frag: row m=l15; lane<16 K in {0..7}u{16..23}, lane>=16 +8 */
  const _Float16* Ap = emb + (size_t)(mt * 16 + l15) * K3H + hi * 8;
  /* B frag: col n=l15; lane<16 K in {0..7}u{16..23}, lane>=16 +8 (from W1^T rows) */
  const _Float16* Bp = w1t + (size_t)head * K3H * HID_ +
                       (size_t)(nt * 16 + l15) * K3H + hi * 8;

  v8f c = {};
#pragma unroll 4
  for (int k0 = 0; k0 < K3H; k0 += 32) {
    v8h a0  = *(const v8h*)(Ap + k0);        /* halves 0..7  */
    v8h a1  = *(const v8h*)(Ap + k0 + 16);   /* halves 8..15 */
    v8h bl  = *(const v8h*)(Bp + k0);
    v8h bh  = *(const v8h*)(Bp + k0 + 16);
    v16h a  = __builtin_shufflevector(a0, a1, 0,1,2,3,4,5,6,7,8,9,10,11,12,13,14,15);
    v16h bb = __builtin_shufflevector(bl, bh, 0,1,2,3,4,5,6,7,8,9,10,11,12,13,14,15);
    c = __builtin_amdgcn_wmma_f32_16x16x32_f16(
        /*neg_a=*/false, a, /*neg_b=*/false, bb,
        /*c_mod=*/(short)0, c, /*reuse_a=*/false, /*reuse_b=*/false);
  }

  const int   n    = nt * 16 + l15;
  const float bias = b1[n];
  float* out = hidden + (size_t)head * ROWS * HID_ + n;
#pragma unroll
  for (int v = 0; v < 8; ++v) {               /* C/D: vgpr v -> row m */
    const int m = mt * 16 + v + hi * 8;
    const float val = c[v] + bias;
    out[(size_t)m * HID_] = val > 0.f ? val : 0.f;
  }
}

/* --------- kernel 3: tiny second GEMMs + softmax/argmax/sigmoid ----------- */
__global__ __launch_bounds__(32) void head_finish(
    const float* __restrict__ hidden,
    const float* __restrict__ mW2, const float* __restrict__ mb2,
    const float* __restrict__ pW2, const float* __restrict__ pb2,
    const float* __restrict__ aW2, const float* __restrict__ ab2,
    const float* __restrict__ iW2, const float* __restrict__ ib2,
    float* __restrict__ out) {
  const int row = blockIdx.x;
  const int t   = threadIdx.x;
  __shared__ float sv[25];
  __shared__ int   spred;

  if (t < 25) {
    int head, col, nc; const float *W2, *b2;
    if (t < 3)       { head = 0; col = t;      nc = 3;  W2 = mW2; b2 = mb2; }
    else if (t < 9)  { head = 1; col = t - 3;  nc = 6;  W2 = pW2; b2 = pb2; }
    else if (t < 21) { head = 2; col = t - 9;  nc = 12; W2 = aW2; b2 = ab2; }
    else             { head = 3; col = t - 21; nc = 4;  W2 = iW2; b2 = ib2; }
    const float* hp = hidden + (size_t)head * ROWS * HID_ + (size_t)row * HID_;
    float acc = b2[col];
    for (int k = 0; k < HID_; ++k) acc += hp[k] * W2[k * nc + col];
    sv[t] = acc;
  }
  __syncthreads();

  if (t == 0) {
    const float l0 = sv[0], l1 = sv[1], l2 = sv[2];
    const float mx = fmaxf(l0, fmaxf(l1, l2));
    const float e0 = expf(l0 - mx), e1 = expf(l1 - mx), e2 = expf(l2 - mx);
    const float inv = 1.f / (e0 + e1 + e2);
    out[row * 3 + 0] = l0; out[row * 3 + 1] = l1; out[row * 3 + 2] = l2;
    out[1536 + row * 3 + 0] = e0 * inv;
    out[1536 + row * 3 + 1] = e1 * inv;
    out[1536 + row * 3 + 2] = e2 * inv;
    spred = (l0 >= l1 && l0 >= l2) ? 0 : (l1 >= l2 ? 1 : 2);   /* first-max */
  }
  __syncthreads();

  if (t < 22) {
    const int pred = spred;
    float v = 0.f;
    if      (pred == 0 && t < 6)              v = 1.f / (1.f + expf(-sv[3 + t]));
    else if (pred == 1 && t >= 6 && t < 18)   v = 1.f / (1.f + expf(-sv[9 + (t - 6)]));
    else if (pred == 2 && t >= 18)            v = 1.f / (1.f + expf(-sv[21 + (t - 18)]));
    out[3072 + row * 22 + t] = v;
  }
}

extern "C" void kernel_launch(void* const* d_in, const int* in_sizes, int n_in,
                              void* d_out, int out_size, void* d_ws, size_t ws_size,
                              hipStream_t stream) {
  (void)in_sizes; (void)n_in; (void)out_size; (void)ws_size;
  const float* seq   = (const float*)d_in[0];
  const int*   starts= (const int*)d_in[1];
  const int*   ends  = (const int*)d_in[2];
  const float* mW1 = (const float*)d_in[3];  const float* mb1 = (const float*)d_in[4];
  const float* mW2 = (const float*)d_in[5];  const float* mb2 = (const float*)d_in[6];
  const float* pW1 = (const float*)d_in[7];  const float* pb1 = (const float*)d_in[8];
  const float* pW2 = (const float*)d_in[9];  const float* pb2 = (const float*)d_in[10];
  const float* aW1 = (const float*)d_in[11]; const float* ab1 = (const float*)d_in[12];
  const float* aW2 = (const float*)d_in[13]; const float* ab2 = (const float*)d_in[14];
  const float* iW1 = (const float*)d_in[15]; const float* ib1 = (const float*)d_in[16];
  const float* iW2 = (const float*)d_in[17]; const float* ib2 = (const float*)d_in[18];

  char* ws = (char*)d_ws;
  _Float16* emb    = (_Float16*)(ws + OFF_EMB);
  _Float16* w1t    = (_Float16*)(ws + OFF_W1T);
  float*    csbuf  = (float*)(ws + OFF_CS);
  float*    hidden = (float*)(ws + OFF_HID);   /* aliases csbuf (CS is dead) */
  float*    out    = (float*)d_out;

  /* 0: convert+transpose W1 heads to f16 (B-fragment friendly: W1T[n,k]) */
  prep_w1_all<<<dim3(K3H, 4), 256, 0, stream>>>(mW1, pW1, aW1, iW1, w1t);
  /* 1a: bandwidth-bound chunk-sum pass (the full 100MB read, 1 add/elem,
         1.57M threads for deep memory-level parallelism) */
  chunk_sums<<<(B_ * NCH * H_) / 256, 256, 0, stream>>>(seq, csbuf);
  /* 1b: assemble span means from register-resident chunk sums + L2 edge rows */
  span_embed2<<<dim3(H_ / 256, B_), 256, 0, stream>>>(seq, csbuf, starts, ends, emb);
  /* 2: 4-head GEMM1 + bias + relu via WMMA (2048 wave-tiles, 8 waves/block) */
  mlp1_wmma<<<256, 256, 0, stream>>>(emb, w1t, mb1, pb1, ab1, ib1, hidden);
  /* 3: tiny GEMM2 heads + softmax/argmax + sigmoid + padded select */
  head_finish<<<ROWS, 32, 0, stream>>>(hidden, mW2, mb2, pW2, pb2,
                                       aW2, ab2, iW2, ib2, out);
}
